// MMD_loss_7438883356786
// MI455X (gfx1250) — matrix operand order repaired and checked by reading, not compiled, not run on hardware
//
#include <hip/hip_runtime.h>
#include <hip/hip_bf16.h>

typedef _Float16 v4h  __attribute__((ext_vector_type(4)));
typedef _Float16 v8h  __attribute__((ext_vector_type(8)));
typedef _Float16 v16h __attribute__((ext_vector_type(16)));
typedef float    v8f  __attribute__((ext_vector_type(8)));

#define N_TOT 4096
#define DIM   2048
#define BHALF 2048
#define KC    64
#define KSTR  72    // LDS row stride in halfs: 144B -> 16B aligned, bank-shifted per row
#define PLANE_ELTS (128 * KSTR)

__device__ __forceinline__ float clean(float x) {
    if (__builtin_isnan(x)) return 0.0f;
    if (__builtin_isinf(x)) return x > 0.0f ? 1.0e4f : -1.0e4f;
    return x;
}

__device__ __forceinline__ const float* row_ptr(const float* __restrict__ s,
                                                const float* __restrict__ t, int r) {
    return (r < BHALF) ? (s + (size_t)r * DIM) : (t + (size_t)(r - BHALF) * DIM);
}

// x = hi + lo split (f16x3 GEMM): hi = rn(x), lo = rn(x - hi)
__device__ __forceinline__ void split4(float4 v, v4h& hi, v4h& lo) {
    float x0 = clean(v.x), x1 = clean(v.y), x2 = clean(v.z), x3 = clean(v.w);
    _Float16 h0 = (_Float16)x0, h1 = (_Float16)x1, h2 = (_Float16)x2, h3 = (_Float16)x3;
    hi = (v4h){h0, h1, h2, h3};
    lo = (v4h){(_Float16)(x0 - (float)h0), (_Float16)(x1 - (float)h1),
               (_Float16)(x2 - (float)h2), (_Float16)(x3 - (float)h3)};
}

__device__ __forceinline__ v16h load_frag2(const _Float16* p0, const _Float16* p1) {
    v8h a = *(const v8h*)p0;
    v8h b = *(const v8h*)p1;
    v16h f;
    #pragma unroll
    for (int i = 0; i < 8; ++i) { f[i] = a[i]; f[i + 8] = b[i]; }
    return f;
}

// CDNA5 async global->LDS copy (ASYNCcnt-tracked, no VGPR round-trip)
__device__ __forceinline__ void async_copy16(unsigned int lds_addr, const void* gptr) {
    asm volatile("global_load_async_to_lds_b128 %0, %1, off"
                 :: "v"(lds_addr), "v"((unsigned long long)(uintptr_t)gptr)
                 : "memory");
}

__device__ __forceinline__ void wait_async0() {
#if __has_builtin(__builtin_amdgcn_s_wait_asynccnt)
    __builtin_amdgcn_s_wait_asynccnt(0);
#else
    asm volatile("s_wait_asynccnt 0x0" ::: "memory");
#endif
}

__device__ __forceinline__ unsigned int lds_off(const void* p) {
    return (unsigned int)(uintptr_t)p;   // LDS aperture: addr[31:0] is the LDS offset
}

// accs[0]=signed kernel sum, accs[1]=linear-MMD fallback, accs[2]=bandwidth
__global__ void init_kernel(double* accs) { accs[0] = 0.0; accs[1] = 0.0; accs[2] = 0.0; }

__global__ void row_sq_kernel(const float* __restrict__ src,
                              const float* __restrict__ tgt,
                              float* __restrict__ sq) {
    int row = blockIdx.x;
    const float* base = row_ptr(src, tgt, row);
    float acc = 0.0f;
    for (int c = threadIdx.x; c < DIM; c += 256) {
        float v = clean(base[c]);
        acc = fmaf(v, v, acc);
    }
    __shared__ float red[256];
    red[threadIdx.x] = acc; __syncthreads();
    for (int s = 128; s > 0; s >>= 1) {
        if (threadIdx.x < (unsigned)s) red[threadIdx.x] += red[threadIdx.x + s];
        __syncthreads();
    }
    if (threadIdx.x == 0) sq[row] = red[0];
}

__global__ void col_kernel(const float* __restrict__ src,
                           const float* __restrict__ tgt,
                           float* __restrict__ colsum, double* accs) {
    int col = blockIdx.x * 256 + threadIdx.x;
    float cs = 0.0f, ct = 0.0f;
    for (int r = 0; r < BHALF; ++r) {
        cs += clean(src[(size_t)r * DIM + col]);
        ct += clean(tgt[(size_t)r * DIM + col]);
    }
    colsum[col] = cs + ct;
    float delta = (cs - ct) * (1.0f / (float)BHALF);
    __shared__ float red[256];
    red[threadIdx.x] = delta * delta; __syncthreads();
    for (int s = 128; s > 0; s >>= 1) {
        if (threadIdx.x < (unsigned)s) red[threadIdx.x] += red[threadIdx.x + s];
        __syncthreads();
    }
    if (threadIdx.x == 0) atomicAdd(&accs[1], (double)red[0]);
}

// bw = max( (2/N)*sum(sq) - (2/N^2)*||colsum||^2 , EPS )  (analytic mean(L2))
__global__ void bw_kernel(const float* __restrict__ sq,
                          const float* __restrict__ colsum, double* accs) {
    double s_sq = 0.0, s_g = 0.0;
    for (int i = threadIdx.x; i < N_TOT; i += 256) s_sq += (double)sq[i];
    for (int i = threadIdx.x; i < DIM; i += 256) { double c = (double)colsum[i]; s_g += c * c; }
    __shared__ double r1[256], r2[256];
    r1[threadIdx.x] = s_sq; r2[threadIdx.x] = s_g; __syncthreads();
    for (int s = 128; s > 0; s >>= 1) {
        if (threadIdx.x < (unsigned)s) {
            r1[threadIdx.x] += r1[threadIdx.x + s];
            r2[threadIdx.x] += r2[threadIdx.x + s];
        }
        __syncthreads();
    }
    if (threadIdx.x == 0) {
        const double N = (double)N_TOT;
        double meanL2 = 2.0 * r1[0] / N - 2.0 * r2[0] / (N * N);
        if (meanL2 < 1.0e-6) meanL2 = 1.0e-6;
        accs[2] = meanL2;
    }
}

// Pre-split the 4096x2048 cleaned matrix into f16 hi/lo planes in workspace.
__global__ void convert_kernel(const float* __restrict__ src, const float* __restrict__ tgt,
                               _Float16* __restrict__ thi, _Float16* __restrict__ tlo) {
    size_t idx = ((size_t)blockIdx.x * 256 + threadIdx.x) * 4;
    int row = (int)(idx >> 11);
    int col = (int)(idx & (DIM - 1));
    float4 v = *(const float4*)(row_ptr(src, tgt, row) + col);
    v4h hi, lo;
    split4(v, hi, lo);
    *(v4h*)(thi + idx) = hi;
    *(v4h*)(tlo + idx) = lo;
}

// Fused Gram via f16x3 WMMA on a lower-triangular block grid, async double-buffered.
template <bool PRE>
__global__ void __launch_bounds__(256)
mmd_gemm_kernel(const float* __restrict__ src, const float* __restrict__ tgt,
                const _Float16* __restrict__ thi, const _Float16* __restrict__ tlo,
                const float* __restrict__ sq, double* accs) {
    // [buf][plane: 0=Ahi 1=Alo 2=Bhi 3=Blo][128*KSTR]
    __shared__ _Float16 smem[2 * 4 * PLANE_ELTS];

    const int t     = threadIdx.x;
    const int lane  = t & 31;
    const int wid   = t >> 5;
    const int waveM = wid >> 1;        // 0..3 -> 32 rows each
    const int waveN = wid & 1;         // 0..1 -> 64 cols each
    const int m16   = lane & 15;
    const int h     = lane >> 4;

    // Triangular block index -> (bm, bn), bn <= bm
    const int b = blockIdx.x;
    int bm = (int)((sqrtf(8.0f * (float)b + 1.0f) - 1.0f) * 0.5f);
    while ((bm + 1) * (bm + 2) / 2 <= b) ++bm;
    while (bm * (bm + 1) / 2 > b) --bm;
    const int bn = b - bm * (bm + 1) / 2;
    const int blockM = bm * 128;
    const int blockN = bn * 128;
    const float wgt  = (bm == bn) ? 1.0f : 2.0f;   // symmetric integrand

    v8f c[2][4];
    #pragma unroll
    for (int ti = 0; ti < 2; ++ti)
        #pragma unroll
        for (int tj = 0; tj < 4; ++tj)
            c[ti][tj] = (v8f){0.f,0.f,0.f,0.f,0.f,0.f,0.f,0.f};

    auto stage = [&](int buf, int k0) {
        _Float16* dAhi = smem + (buf * 4 + 0) * PLANE_ELTS;
        _Float16* dAlo = smem + (buf * 4 + 1) * PLANE_ELTS;
        _Float16* dBhi = smem + (buf * 4 + 2) * PLANE_ELTS;
        _Float16* dBlo = smem + (buf * 4 + 3) * PLANE_ELTS;
        if (PRE) {
            #pragma unroll
            for (int it = 0; it < 4; ++it) {
                int q   = t + 256 * it;       // 1024 chunks of 8 halfs per plane
                int row = q >> 3;
                int col = (q & 7) * 8;
                size_t ga = (size_t)(blockM + row) * DIM + k0 + col;
                size_t gb = (size_t)(blockN + row) * DIM + k0 + col;
                unsigned int la = (unsigned int)(row * KSTR + col) * 2u;
                async_copy16(lds_off(dAhi) + la, thi + ga);
                async_copy16(lds_off(dAlo) + la, tlo + ga);
                async_copy16(lds_off(dBhi) + la, thi + gb);
                async_copy16(lds_off(dBlo) + la, tlo + gb);
            }
        } else {
            #pragma unroll
            for (int it = 0; it < 8; ++it) {
                int q   = t + 256 * it;       // 2048 float4 chunks per matrix
                int row = q >> 4;
                int col = (q & 15) * 4;
                float4 va = *(const float4*)(row_ptr(src, tgt, blockM + row) + k0 + col);
                float4 vb = *(const float4*)(row_ptr(src, tgt, blockN + row) + k0 + col);
                v4h hi, lo;
                split4(va, hi, lo);
                *(v4h*)&dAhi[row * KSTR + col] = hi;
                *(v4h*)&dAlo[row * KSTR + col] = lo;
                split4(vb, hi, lo);
                *(v4h*)&dBhi[row * KSTR + col] = hi;
                *(v4h*)&dBlo[row * KSTR + col] = lo;
            }
        }
    };

    stage(0, 0);

    const int NCHUNK = DIM / KC;       // 32
    for (int ch = 0; ch < NCHUNK; ++ch) {
        const int buf = ch & 1;
        if (PRE) wait_async0();        // our async copies for this buffer are done
        __syncthreads();               // everyone's are; prev compute on other buf done
        if (ch + 1 < NCHUNK) stage(buf ^ 1, (ch + 1) * KC);

        const _Float16* sAhi = smem + (buf * 4 + 0) * PLANE_ELTS;
        const _Float16* sAlo = smem + (buf * 4 + 1) * PLANE_ELTS;
        const _Float16* sBhi = smem + (buf * 4 + 2) * PLANE_ELTS;
        const _Float16* sBlo = smem + (buf * 4 + 3) * PLANE_ELTS;

        #pragma unroll
        for (int kg = 0; kg < KC / 32; ++kg) {
            // A fragments (16x32 f16): lane holds K in [8h,8h+8) u [16+8h,16+8h+8)
            v16h ahi[2], alo[2];
            #pragma unroll
            for (int ti = 0; ti < 2; ++ti) {
                int row = waveM * 32 + ti * 16 + m16;
                const _Float16* bhp = &sAhi[row * KSTR + kg * 32];
                const _Float16* blp = &sAlo[row * KSTR + kg * 32];
                ahi[ti] = load_frag2(bhp + 8 * h, bhp + 16 + 8 * h);
                alo[ti] = load_frag2(blp + 8 * h, blp + 16 + 8 * h);
            }
            #pragma unroll
            for (int tj = 0; tj < 4; ++tj) {
                // B fragments (32x16 f16): lane holds K in [8h,8h+8) u [16+8h,16+8h+8)
                int row = waveN * 64 + tj * 16 + m16;
                const _Float16* bhp = &sBhi[row * KSTR + kg * 32];
                const _Float16* blp = &sBlo[row * KSTR + kg * 32];
                v16h bhi = load_frag2(bhp + 8 * h, bhp + 16 + 8 * h);
                v16h blo = load_frag2(blp + 8 * h, blp + 16 + 8 * h);
                #pragma unroll
                for (int ti = 0; ti < 2; ++ti) {
                    c[ti][tj] = __builtin_amdgcn_wmma_f32_16x16x32_f16(
                        false, ahi[ti], false, bhi, (short)0, c[ti][tj], false, false);
                    c[ti][tj] = __builtin_amdgcn_wmma_f32_16x16x32_f16(
                        false, ahi[ti], false, blo, (short)0, c[ti][tj], false, false);
                    c[ti][tj] = __builtin_amdgcn_wmma_f32_16x16x32_f16(
                        false, alo[ti], false, bhi, (short)0, c[ti][tj], false, false);
                }
            }
        }
    }

    // Epilogue: L2 -> 5-scale RBF kernels -> signed triangular-weighted sum
    const float bwf = (float)accs[2];
    float inv[5];
    {
        float bcur = bwf;
        #pragma unroll
        for (int i = 0; i < 5; ++i) {
            float bi = fmaxf(bcur, 1.0e-6f);
            inv[i] = 1.0f / bi;
            bcur *= 2.0f;
        }
    }

    float acc = 0.0f;
    #pragma unroll
    for (int ti = 0; ti < 2; ++ti) {
        #pragma unroll
        for (int tj = 0; tj < 4; ++tj) {
            #pragma unroll
            for (int r = 0; r < 8; ++r) {
                int gi = blockM + waveM * 32 + ti * 16 + r + 8 * h;   // C row (M)
                int gj = blockN + waveN * 64 + tj * 16 + m16;         // C col (N)
                float l2 = fmaxf(sq[gi] + sq[gj] - 2.0f * c[ti][tj][r], 0.0f);
                float ks = 0.0f;
                #pragma unroll
                for (int i = 0; i < 5; ++i)
                    ks += __expf(-fminf(l2 * inv[i], 50.0f));
                float sgn = ((gi < BHALF) == (gj < BHALF)) ? 1.0f : -1.0f;
                acc += sgn * ks;
            }
        }
    }

    __shared__ float red[256];
    red[t] = acc; __syncthreads();
    for (int s = 128; s > 0; s >>= 1) {
        if (t < s) red[t] += red[t + s];
        __syncthreads();
    }
    if (t == 0) atomicAdd(&accs[0], (double)(red[0] * wgt));
}

__global__ void finalize_kernel(const double* accs, float* out) {
    double loss = accs[0] / ((double)BHALF * (double)BHALF);
    float f = (float)loss;
    out[0] = __builtin_isfinite(f) ? f : (float)accs[1];
}

extern "C" void kernel_launch(void* const* d_in, const int* in_sizes, int n_in,
                              void* d_out, int out_size, void* d_ws, size_t ws_size,
                              hipStream_t stream) {
    const float* src = (const float*)d_in[0];
    const float* tgt = (const float*)d_in[1];
    float* out = (float*)d_out;

    const size_t plane = (size_t)N_TOT * DIM * sizeof(_Float16);  // 16 MB per f16 plane
    const bool pre = ws_size >= 2 * plane + 32768;

    char* base = (char*)d_ws;
    _Float16 *thi = nullptr, *tlo = nullptr;
    float *sq, *colsum;
    double* accs;
    if (pre) {
        thi    = (_Float16*)base;
        tlo    = (_Float16*)(base + plane);
        sq     = (float*)(base + 2 * plane);
        colsum = sq + N_TOT;
        accs   = (double*)(base + 2 * plane + 24576);
    } else {
        sq     = (float*)base;
        colsum = sq + N_TOT;
        accs   = (double*)(base + 24576);
    }

    init_kernel<<<1, 1, 0, stream>>>(accs);
    row_sq_kernel<<<N_TOT, 256, 0, stream>>>(src, tgt, sq);
    col_kernel<<<DIM / 256, 256, 0, stream>>>(src, tgt, colsum, accs);
    bw_kernel<<<1, 256, 0, stream>>>(sq, colsum, accs);

    const int ntiles = N_TOT / 128;                       // 32
    const int ntri   = ntiles * (ntiles + 1) / 2;         // 528
    if (pre) {
        convert_kernel<<<(N_TOT * DIM / 4) / 256, 256, 0, stream>>>(src, tgt, thi, tlo);
        mmd_gemm_kernel<true><<<ntri, 256, 0, stream>>>(src, tgt, thi, tlo, sq, accs);
    } else {
        mmd_gemm_kernel<false><<<ntri, 256, 0, stream>>>(src, tgt, nullptr, nullptr, sq, accs);
    }
    finalize_kernel<<<1, 1, 0, stream>>>(accs, out);
}
